// DotModule_27066883899823
// MI455X (gfx1250) — 8-bit matrix operand order repaired and checked by reading, compiled, not run on hardware
//
#include <hip/hip_runtime.h>
#include <cstdint>
#include <cstddef>

typedef __attribute__((ext_vector_type(8))) int v8i;

#define N_MAT 4096

__global__ void absmax_kernel(const float* __restrict__ x, unsigned* __restrict__ out, int n4) {
  __shared__ unsigned red[256];
  unsigned m = 0u;
  const float4* __restrict__ x4 = (const float4*)x;
  for (int i = blockIdx.x * blockDim.x + threadIdx.x; i < n4; i += gridDim.x * blockDim.x) {
    float4 v = x4[i];
    unsigned a = __float_as_uint(fabsf(v.x));
    unsigned b = __float_as_uint(fabsf(v.y));
    unsigned c = __float_as_uint(fabsf(v.z));
    unsigned d = __float_as_uint(fabsf(v.w));
    a = a > b ? a : b;
    c = c > d ? c : d;
    a = a > c ? a : c;
    m = m > a ? m : a;
  }
  red[threadIdx.x] = m;
  __syncthreads();
  for (int s = 128; s > 0; s >>= 1) {
    if ((int)threadIdx.x < s) {
      unsigned o = red[threadIdx.x + s];
      if (o > red[threadIdx.x]) red[threadIdx.x] = o;
    }
    __syncthreads();
  }
  if (threadIdx.x == 0) atomicMax(out, red[0]);
}

__device__ __forceinline__ signed char quant1(float v, float s) {
  float q = rintf(v * s);
  q = fminf(fmaxf(q, -127.0f), 127.0f);
  return (signed char)(int)q;
}

__global__ void quant_rm_kernel(const float* __restrict__ x, signed char* __restrict__ q,
                                const unsigned* __restrict__ maxbits, int n4) {
  const float bound = fmaxf(__uint_as_float(*maxbits), 1e-6f);
  const float s = 127.0f / bound;
  const float4* __restrict__ x4 = (const float4*)x;
  char4* __restrict__ q4 = (char4*)q;
  for (int i = blockIdx.x * blockDim.x + threadIdx.x; i < n4; i += gridDim.x * blockDim.x) {
    float4 v = x4[i];
    char4 o;
    o.x = quant1(v.x, s);
    o.y = quant1(v.y, s);
    o.z = quant1(v.z, s);
    o.w = quant1(v.w, s);
    q4[i] = o;
  }
}

__global__ void quant_tr_kernel(const float* __restrict__ x, signed char* __restrict__ qt,
                                const unsigned* __restrict__ maxbits) {
  __shared__ signed char tile[32][33];
  const float bound = fmaxf(__uint_as_float(*maxbits), 1e-6f);
  const float s = 127.0f / bound;
  const int n0 = blockIdx.x * 32;
  const int k0 = blockIdx.y * 32;
  for (int r = threadIdx.y; r < 32; r += 8) {
    tile[r][threadIdx.x] = quant1(x[(size_t)(k0 + r) * N_MAT + n0 + threadIdx.x], s);
  }
  __syncthreads();
  for (int r = threadIdx.y; r < 32; r += 8) {
    qt[(size_t)(n0 + r) * N_MAT + k0 + threadIdx.x] = tile[threadIdx.x][r];
  }
}

__device__ __forceinline__ void async_copy_b128(unsigned lds_addr, const void* gaddr) {
  asm volatile("global_load_async_to_lds_b128 %0, %1, off"
               :: "v"(lds_addr), "v"((unsigned long long)(uintptr_t)gaddr)
               : "memory");
}
__device__ __forceinline__ void wait_asynccnt0() {
  asm volatile("s_wait_asynccnt 0x0" ::: "memory");
}

__global__ __launch_bounds__(128) void gemm_iu8_kernel(
    const signed char* __restrict__ qA,
    const signed char* __restrict__ qBT,
    float* __restrict__ out,
    const unsigned* __restrict__ scalebits) {
  __shared__ alignas(16) signed char As[2][128][64];
  __shared__ alignas(16) signed char Bs[2][128][64];

  const int tid  = threadIdx.x;
  const int lane = tid & 31;
  const int wave = tid >> 5;
  const int waveM = (wave >> 1) * 64;
  const int waveN = (wave & 1) * 64;
  const int blockM = blockIdx.y * 128;
  const int blockN = blockIdx.x * 128;

  v8i acc[4][4];
#pragma unroll
  for (int i = 0; i < 4; ++i)
#pragma unroll
    for (int j = 0; j < 4; ++j) {
      v8i z = {0, 0, 0, 0, 0, 0, 0, 0};
      acc[i][j] = z;
    }

  const int lrow = tid >> 2;
  const int lcol = (tid & 3) * 16;

  unsigned aLds[4], bLds[4];
  const signed char* aPtr[4];
  const signed char* bPtr[4];
#pragma unroll
  for (int r = 0; r < 4; ++r) {
    aLds[r] = (unsigned)(uintptr_t)&As[0][r * 32 + lrow][lcol];
    bLds[r] = (unsigned)(uintptr_t)&Bs[0][r * 32 + lrow][lcol];
    aPtr[r] = qA  + (size_t)(blockM + r * 32 + lrow) * N_MAT + lcol;
    bPtr[r] = qBT + (size_t)(blockN + r * 32 + lrow) * N_MAT + lcol;
  }

#pragma unroll
  for (int r = 0; r < 4; ++r) {
    async_copy_b128(aLds[r], aPtr[r]);
    async_copy_b128(bLds[r], bPtr[r]);
  }

  const int mrow  = lane & 15;
  const int koffA = (lane >> 4) * 8;
  const int koffB = (lane >> 4) * 8;

  const int KSTEPS = N_MAT / 64;
  for (int step = 0; step < KSTEPS; ++step) {
    const int buf = step & 1;

    wait_asynccnt0();
    __syncthreads();

    if (step + 1 < KSTEPS) {
      const int k0 = (step + 1) * 64;
      const unsigned ldsOff = (unsigned)((buf ^ 1) * 8192);
#pragma unroll
      for (int r = 0; r < 4; ++r) {
        async_copy_b128(aLds[r] + ldsOff, aPtr[r] + k0);
        async_copy_b128(bLds[r] + ldsOff, bPtr[r] + k0);
      }
    }

    v8i afrag[4];
#pragma unroll
    for (int t = 0; t < 4; ++t) {
      const signed char* p = &As[buf][waveM + t * 16 + mrow][koffA];
      int2 c0 = *(const int2*)(p + 0);
      int2 c1 = *(const int2*)(p + 16);
      int2 c2 = *(const int2*)(p + 32);
      int2 c3 = *(const int2*)(p + 48);
      v8i a = {c0.x, c0.y, c1.x, c1.y, c2.x, c2.y, c3.x, c3.y};
      afrag[t] = a;
    }
#pragma unroll
    for (int j = 0; j < 4; ++j) {
      const signed char* p = &Bs[buf][waveN + j * 16 + mrow][koffB];
      int2 d0 = *(const int2*)(p + 0);
      int2 d1 = *(const int2*)(p + 16);
      int2 d2 = *(const int2*)(p + 32);
      int2 d3 = *(const int2*)(p + 48);
      v8i b = {d0.x, d0.y, d1.x, d1.y, d2.x, d2.y, d3.x, d3.y};
#pragma unroll
      for (int i = 0; i < 4; ++i) {
        acc[i][j] = __builtin_amdgcn_wmma_i32_16x16x64_iu8(
            true, afrag[i], true, b, acc[i][j], false, false);
      }
    }
  }

  const float bl = fmaxf(__uint_as_float(scalebits[0]), 1e-6f);
  const float br = fmaxf(__uint_as_float(scalebits[1]), 1e-6f);
  const float dq = (bl * br) * (1.0f / (127.0f * 127.0f));

  const int nlo = lane & 15;
  const int mhi = (lane >> 4) * 8;
#pragma unroll
  for (int i = 0; i < 4; ++i) {
#pragma unroll
    for (int j = 0; j < 4; ++j) {
      const int row0 = blockM + waveM + i * 16 + mhi;
      const int col  = blockN + waveN + j * 16 + nlo;
#pragma unroll
      for (int r = 0; r < 8; ++r) {
        out[(size_t)(row0 + r) * N_MAT + col] = (float)acc[i][j][r] * dq;
      }
    }
  }
}

extern "C" void kernel_launch(void* const* d_in, const int* in_sizes, int n_in,
                              void* d_out, int out_size, void* d_ws, size_t ws_size,
                              hipStream_t stream) {
  (void)in_sizes; (void)n_in; (void)out_size; (void)ws_size;

  const float* lhs = (const float*)d_in[0];
  const float* rhs = (const float*)d_in[1];
  float* out = (float*)d_out;

  unsigned* scalebits = (unsigned*)d_ws;
  signed char* qA  = (signed char*)d_ws + 256;
  signed char* qBT = qA + (size_t)N_MAT * N_MAT;

  hipMemsetAsync(d_ws, 0, 256, stream);

  const int n4 = (N_MAT * N_MAT) / 4;
  absmax_kernel<<<512, 256, 0, stream>>>(lhs, scalebits + 0, n4);
  absmax_kernel<<<512, 256, 0, stream>>>(rhs, scalebits + 1, n4);

  quant_rm_kernel<<<2048, 256, 0, stream>>>(lhs, qA, scalebits + 0, n4);
  quant_tr_kernel<<<dim3(N_MAT / 32, N_MAT / 32), dim3(32, 8), 0, stream>>>(rhs, qBT, scalebits + 1);

  gemm_iu8_kernel<<<dim3(N_MAT / 128, N_MAT / 128), 128, 0, stream>>>(qA, qBT, out, scalebits);
}
